// GraphEncoder_59785944760593
// MI455X (gfx1250) — matrix operand order repaired and checked by reading, not compiled, not run on hardware
//
#include <hip/hip_runtime.h>

#define DFEAT  128
#define NHEADS 8
#define FPH    16   // features per head

typedef __attribute__((ext_vector_type(16))) _Float16 v16h;
typedef __attribute__((ext_vector_type(8)))  float    v8f;

// ---------------------------------------------------------------------------
// h = X @ W  (N x 128) = (N x 128)(128 x 128), fp32 in/out.
// f16 WMMA with hi/lo error-compensated split: X*W ~= Xh*Wh + Xh*Wl + Xl*Wh
// (fp32 accumulate -> ~21 effective mantissa bits).
// One wave per 16x16 output tile; blockDim = (32,8): wave y = column tile.
// K-loop fully unrolled: 12 v_wmma per tile.
// ---------------------------------------------------------------------------
__global__ __launch_bounds__(256) void gat_gemm_wmma(
    const float* __restrict__ X, const float* __restrict__ W,
    float* __restrict__ Hout, int N)
{
    const int lane = threadIdx.x;       // 0..31
    const int wv   = threadIdx.y;       // 0..7  -> column tile
    const int half = lane >> 4;
    const int mr   = lane & 15;
    const int m0   = blockIdx.x * 16;
    const int n0   = wv * 16;
    const int row  = m0 + mr;
    const bool fullTile = (m0 + 16 <= N);
    const int rowc = (row < N) ? row : 0;   // clamp, value masked below

    v8f acc = {};
#pragma unroll
    for (int ks = 0; ks < DFEAT; ks += 32) {
        // ---- A tile (16x32 of X), ISA A-layout: VGPR0-3: K=half*8+0..7,
        //      VGPR4-7: K=16+half*8+0..7 ----
        const float* xr = X + (size_t)rowc * DFEAT + ks + half * 8;
        float av[16];
#pragma unroll
        for (int j = 0; j < 8; ++j) av[j]     = xr[j];
#pragma unroll
        for (int j = 0; j < 8; ++j) av[8 + j] = xr[16 + j];

        v16h a_hi, a_lo;
#pragma unroll
        for (int e = 0; e < 16; ++e) {
            float v = (row < N) ? av[e] : 0.0f;
            _Float16 hi = (_Float16)v;
            a_hi[e] = hi;
            a_lo[e] = (_Float16)(v - (float)hi);
        }

        // ---- B tile (32x16 of W), ISA B-layout: lane n = mr, element e
        //      holds K = half*8 + e (+8 for e>=8); loads coalesce across the half-wave ----
        v16h b_hi, b_lo;
#pragma unroll
        for (int e = 0; e < 16; ++e) {
            float v = W[(size_t)(ks + half * 8 + e + (e >= 8 ? 8 : 0)) * DFEAT + n0 + mr];
            _Float16 hi = (_Float16)v;
            b_hi[e] = hi;
            b_lo[e] = (_Float16)(v - (float)hi);
        }

        acc = __builtin_amdgcn_wmma_f32_16x16x32_f16(false, a_hi, false, b_hi,
                                                     (short)0, acc, false, false);
        acc = __builtin_amdgcn_wmma_f32_16x16x32_f16(false, a_hi, false, b_lo,
                                                     (short)0, acc, false, false);
        acc = __builtin_amdgcn_wmma_f32_16x16x32_f16(false, a_lo, false, b_hi,
                                                     (short)0, acc, false, false);
    }

    // C/D layout: VGPR r -> M = r + 8*half, N = mr.
    // Fast path (all tiles when N % 16 == 0): branch-free stores, the
    // r*DFEAT stride folds into the instruction's immediate offset.
    float* op = Hout + (size_t)(m0 + 8 * half) * DFEAT + n0 + mr;
    if (fullTile) {
#pragma unroll
        for (int r = 0; r < 8; ++r) op[(size_t)r * DFEAT] = acc[r];
    } else {
#pragma unroll
        for (int r = 0; r < 8; ++r) {
            if (m0 + 8 * half + r < N) op[(size_t)r * DFEAT] = acc[r];
        }
    }
}

// ---------------------------------------------------------------------------
// Per (node, head): attention scores; also init amax=-inf, asum=0.
// ---------------------------------------------------------------------------
__global__ void gat_node_scores(const float* __restrict__ Hf,
                                const float* __restrict__ att_src,
                                const float* __restrict__ att_dst,
                                float* __restrict__ a_src, float* __restrict__ a_dst,
                                float* __restrict__ amax, float* __restrict__ asum,
                                int N)
{
    int t = blockIdx.x * blockDim.x + threadIdx.x;
    if (t >= N * NHEADS) return;
    int n = t >> 3, hh = t & 7;
    const float* hp = Hf + (size_t)n * DFEAT + hh * FPH;
    const float* as = att_src + hh * FPH;
    const float* ad = att_dst + hh * FPH;
    float s = 0.0f, d = 0.0f;
#pragma unroll
    for (int f = 0; f < FPH; ++f) { float v = hp[f]; s += v * as[f]; d += v * ad[f]; }
    a_src[t] = s;
    a_dst[t] = d;
    amax[t]  = -__builtin_inff();
    asum[t]  = 0.0f;
}

// out[n][d] = bias[d]  (bias folded into the segment-sum accumulator)
__global__ void gat_init_out(float* __restrict__ out, const float* __restrict__ bias,
                             int total)
{
    int t = blockIdx.x * blockDim.x + threadIdx.x;
    if (t < total) out[t] = bias[t & (DFEAT - 1)];
}

// float atomic-max via signed/unsigned int ordering trick
__device__ inline void atomicMaxF(float* addr, float val)
{
    if (val >= 0.0f) atomicMax((int*)addr, __float_as_int(val));
    else             atomicMin((unsigned int*)addr, __float_as_uint(val));
}

// Pass 1: alpha = leaky_relu(a_src[src]+a_dst[dst]); amax = segment max over dst
__global__ void gat_edge_alpha(const int* __restrict__ ei, int E, int N,
                               const float* __restrict__ a_src,
                               const float* __restrict__ a_dst,
                               float* __restrict__ alpha, float* __restrict__ amax)
{
    int t = blockIdx.x * blockDim.x + threadIdx.x;
    int e = t >> 3, hh = t & 7;
    if (e >= E + N) return;
    int s, d;
    if (e < E) { s = ei[e]; d = ei[E + e]; } else { s = d = e - E; }  // self-loops
    float v = a_src[s * NHEADS + hh] + a_dst[d * NHEADS + hh];
    v = (v > 0.0f) ? v : 0.2f * v;
    alpha[t] = v;
    atomicMaxF(&amax[d * NHEADS + hh], v);
}

// Pass 2: alpha = exp(alpha - amax[dst]); asum = segment sum over dst
__global__ void gat_edge_exp(const int* __restrict__ ei, int E, int N,
                             float* __restrict__ alpha,
                             const float* __restrict__ amax,
                             float* __restrict__ asum)
{
    int t = blockIdx.x * blockDim.x + threadIdx.x;
    int e = t >> 3, hh = t & 7;
    if (e >= E + N) return;
    int d = (e < E) ? ei[E + e] : (e - E);
    float w = __expf(alpha[t] - amax[d * NHEADS + hh]);
    alpha[t] = w;
    atomicAdd(&asum[d * NHEADS + hh], w);
}

// Pass 3: out[dst] += h[src] * alpha / asum[dst].  One thread per 4 features
// (32 threads per edge); h/out/alpha are L2-resident on MI455X (192MB L2),
// so the random gathers + atomic scatters stay on-chip.
__global__ void gat_edge_scatter(const int* __restrict__ ei, int E, int N,
                                 const float* __restrict__ Hf,
                                 const float* __restrict__ alpha,
                                 const float* __restrict__ asum,
                                 float* __restrict__ out)
{
    int t = blockIdx.x * blockDim.x + threadIdx.x;
    int e = t >> 5, q = t & 31;          // q*4 .. q*4+3 features
    if (e >= E + N) return;
    int s, d;
    if (e < E) { s = ei[e]; d = ei[E + e]; } else { s = d = e - E; }
    int hh = q >> 2;                     // head = (q*4)/16
    float coef = alpha[e * NHEADS + hh] / (asum[d * NHEADS + hh] + 1e-16f);
    const float4 hv = *(const float4*)(Hf + (size_t)s * DFEAT + q * 4);
    float* op = out + (size_t)d * DFEAT + q * 4;
    atomicAdd(op + 0, hv.x * coef);
    atomicAdd(op + 1, hv.y * coef);
    atomicAdd(op + 2, hv.z * coef);
    atomicAdd(op + 3, hv.w * coef);
}

// ---------------------------------------------------------------------------
extern "C" void kernel_launch(void* const* d_in, const int* in_sizes, int n_in,
                              void* d_out, int out_size, void* d_ws, size_t ws_size,
                              hipStream_t stream)
{
    const float* x       = (const float*)d_in[0];
    const int*   ei      = (const int*)  d_in[1];
    const float* W       = (const float*)d_in[2];
    const float* att_src = (const float*)d_in[3];
    const float* att_dst = (const float*)d_in[4];
    const float* bias    = (const float*)d_in[5];

    const int N    = in_sizes[0] / DFEAT;   // 100000
    const int E    = in_sizes[1] / 2;       // 1000000
    const int Etot = E + N;                 // + self-loops

    float* ws = (float*)d_ws;
    size_t o = 0;
    float* hbuf  = ws + o; o += (size_t)N * DFEAT;     // 51.2 MB
    float* a_src = ws + o; o += (size_t)N * NHEADS;
    float* a_dst = ws + o; o += (size_t)N * NHEADS;
    float* amax  = ws + o; o += (size_t)N * NHEADS;
    float* asum  = ws + o; o += (size_t)N * NHEADS;
    float* alpha = ws + o; o += (size_t)Etot * NHEADS; // 35.2 MB
    float* x1    = ws + o; o += (size_t)N * DFEAT;     // 51.2 MB

    const float* xin = x;
    for (int layer = 0; layer < 2; ++layer) {
        float* outb = (layer == 0) ? x1 : (float*)d_out;

        dim3 blk(32, 8);
        gat_gemm_wmma<<<(N + 15) / 16, blk, 0, stream>>>(xin, W, hbuf, N);

        int nh = N * NHEADS;
        gat_node_scores<<<(nh + 255) / 256, 256, 0, stream>>>(
            hbuf, att_src, att_dst, a_src, a_dst, amax, asum, N);

        int nd = N * DFEAT;
        gat_init_out<<<(nd + 255) / 256, 256, 0, stream>>>(outb, bias, nd);

        int eh = Etot * NHEADS;
        gat_edge_alpha<<<(eh + 255) / 256, 256, 0, stream>>>(
            ei, E, N, a_src, a_dst, alpha, amax);
        gat_edge_exp<<<(eh + 255) / 256, 256, 0, stream>>>(
            ei, E, N, alpha, amax, asum);

        int et = Etot * 32;
        gat_edge_scatter<<<(et + 255) / 256, 256, 0, stream>>>(
            ei, E, N, hbuf, alpha, asum, outb);

        xin = outb;
    }
}
